// HighSamplingEdgeMatchingNet_57526791962717
// MI455X (gfx1250) — matrix operand order repaired and checked by reading, not compiled, not run on hardware
//
#include <hip/hip_runtime.h>
#include <hip/hip_bf16.h>
#include <math.h>

typedef __attribute__((ext_vector_type(16))) _Float16 v16h;
typedef __attribute__((ext_vector_type(8)))  _Float16 v8h;
typedef __attribute__((ext_vector_type(8)))  float    v8f;

#define B_    32
#define N_    16384
#define SEG   512
#define S_    32
#define FD    128
#define VALID (N_ - SEG)   /* 15872 = 256*62 */
#define SB    (S_ * B_)    /* 1024 */

// position-major LDS row strides (halves); multiples of 16 halves keep the
// 16-half B-fragment runs 32B-aligned and spread lanes over 8 bank groups.
#define H1S   80    /* 64 channels + 16 pad */
#define H2S   144   /* 128 channels + 16 pad */

__device__ __forceinline__ float hash_u01(unsigned s) {
  s ^= s >> 16; s *= 0x7feb352dU;
  s ^= s >> 15; s *= 0x846ca68bU;
  s ^= s >> 16;
  return ((float)(s >> 8) + 0.5f) * (1.0f / 16777216.0f);
}

// ---------------------------------------------------------------------------
// K1: importance net (conv 2->64 k5, 64->32 k3, 32->1 k1, sigmoid) + curvature
// one block per (tile of 512 L, batch, which). Fused in LDS.
// ---------------------------------------------------------------------------
__global__ __launch_bounds__(256) void imp_comb_kernel(
    const float* __restrict__ pts1, const float* __restrict__ pts2,
    const float* __restrict__ w1, const float* __restrict__ b1,
    const float* __restrict__ w2, const float* __restrict__ b2,
    const float* __restrict__ w3, const float* __restrict__ b3,
    float* __restrict__ comb1, float* __restrict__ comb2)
{
  extern __shared__ unsigned char smem[];
  float* xs  = (float*)smem;        // 2*518
  float* w1l = xs + 2 * 518;        // 640
  float* b1l = w1l + 640;           // 64
  float* w2l = b1l + 64;            // 6144
  float* b2l = w2l + 6144;          // 32
  float* w3l = b2l + 32;            // 32
  _Float16* h1 = (_Float16*)(w3l + 32);  // 64*514 (l in [t0-1, t0+512])

  const int tid   = threadIdx.x;
  const int tile  = blockIdx.x;     // 0..31
  const int b     = blockIdx.y;     // 0..31
  const int which = blockIdx.z;     // 0..1
  const float* pts  = which ? pts2 : pts1;
  float*       comb = which ? comb2 : comb1;
  const int t0 = tile * 512;

  for (int i = tid; i < 640;  i += 256) w1l[i] = w1[i];
  for (int i = tid; i < 64;   i += 256) b1l[i] = b1[i];
  for (int i = tid; i < 6144; i += 256) w2l[i] = w2[i];
  if (tid < 32) { b2l[tid] = b2[tid]; w3l[tid] = w3[tid]; }
  for (int i = tid; i < 2 * 518; i += 256) {
    int ci = i / 518, p = i % 518;
    int l = t0 - 3 + p;  // x needed on [t0-3, t0+514]
    xs[ci * 518 + p] = (l >= 0 && l < N_) ? pts[((size_t)b * N_ + l) * 2 + ci] : 0.0f;
  }
  __syncthreads();

  // conv1 + relu over l in [t0-1, t0+512] (zero outside [0,N))
  for (int o = tid; o < 64 * 514; o += 256) {
    int co = o / 514, q = o % 514;
    int l = t0 - 1 + q;
    float acc = 0.0f;
    if (l >= 0 && l < N_) {
      acc = b1l[co];
      #pragma unroll
      for (int ci = 0; ci < 2; ++ci)
        #pragma unroll
        for (int t = 0; t < 5; ++t)
          acc += w1l[(co * 2 + ci) * 5 + t] * xs[ci * 518 + q + t];
      acc = fmaxf(acc, 0.0f);
    }
    h1[co * 514 + q] = (_Float16)acc;
  }
  __syncthreads();

  // conv2 + relu, conv3 (1x1) + sigmoid, curvature, comb
  const float b3v = b3[0];
  for (int l0 = tid; l0 < 512; l0 += 256) {
    int l = t0 + l0;
    float out = b3v;
    for (int c = 0; c < 32; ++c) {
      float acc = b2l[c];
      #pragma unroll 4
      for (int ci = 0; ci < 64; ++ci) {
        const float*    wr = &w2l[(c * 64 + ci) * 3];
        const _Float16* hr = &h1[ci * 514 + l0];
        acc += wr[0] * (float)hr[0] + wr[1] * (float)hr[1] + wr[2] * (float)hr[2];
      }
      out += w3l[c] * fmaxf(acc, 0.0f);
    }
    float imp = 1.0f / (1.0f + __expf(-out));
    int lp = (l + 1) % N_;
    int lm = (l + N_ - 1) % N_;
    float px = pts[((size_t)b * N_ + l) * 2 + 0], py = pts[((size_t)b * N_ + l) * 2 + 1];
    float fx = pts[((size_t)b * N_ + lp) * 2 + 0] - px;
    float fy = pts[((size_t)b * N_ + lp) * 2 + 1] - py;
    float bx = px - pts[((size_t)b * N_ + lm) * 2 + 0];
    float by = py - pts[((size_t)b * N_ + lm) * 2 + 1];
    comb[(size_t)b * N_ + l] = imp + 0.3f * fabsf(fx * by - fy * bx);
  }
}

// ---------------------------------------------------------------------------
// K2: sequential multinomial sampling with importance decay.
// ---------------------------------------------------------------------------
__global__ __launch_bounds__(256) void sampling_kernel(
    const float* __restrict__ comb1, const float* __restrict__ comb2,
    int* __restrict__ st1, int* __restrict__ st2)
{
  extern __shared__ unsigned char smem[];
  float* cl  = (float*)smem;  // N_
  float* red = cl + N_;       // 256
  float* bc  = red + 256;     // broadcast slots
  int*   ibc = (int*)bc;

  const int tid = threadIdx.x;
  const int b = blockIdx.x;
  const int which = blockIdx.y;
  const float* comb = which ? comb2 : comb1;
  int* st = which ? st2 : st1;

  for (int i = tid; i < N_; i += 256) cl[i] = comb[(size_t)b * N_ + i];
  __syncthreads();

  const int CH = VALID / 256;  // 62
  const int base = tid * CH;
  for (int s = 0; s < S_; ++s) {
    float m = -1e30f;
    for (int i = 0; i < CH; ++i) m = fmaxf(m, cl[base + i]);
    red[tid] = m;
    __syncthreads();
    if (tid == 0) {
      float gm = -1e30f;
      for (int i = 0; i < 256; ++i) gm = fmaxf(gm, red[i]);
      bc[1] = gm;
    }
    __syncthreads();
    float gm = bc[1];
    float cs = 0.0f;
    for (int i = 0; i < CH; ++i) cs += __expf(cl[base + i] - gm);
    red[tid] = cs;
    __syncthreads();
    if (tid == 0) {
      float tot = 0.0f;
      for (int i = 0; i < 256; ++i) tot += red[i];
      unsigned seed = 0x9E3779B9u ^ (unsigned)(s * 1315423911u)
                    ^ (unsigned)(b * 2654435761u) ^ (unsigned)(which * 0x85EBCA6Bu);
      float target = hash_u01(seed) * tot;
      float acc = 0.0f;
      int chunk = 255;
      for (int i = 0; i < 256; ++i) {
        float na = acc + red[i];
        if (na >= target) { chunk = i; break; }
        acc = na;
      }
      int idx = chunk * CH + CH - 1;
      for (int i = 0; i < CH; ++i) {
        acc += __expf(cl[chunk * CH + i] - gm);
        if (acc >= target) { idx = chunk * CH + i; break; }
      }
      if (idx >= VALID) idx = VALID - 1;
      ibc[0] = idx;
      st[s * B_ + b] = idx;
    }
    __syncthreads();
    int start = ibc[0];
    for (int i = tid; i < SEG; i += 256) cl[start + i] *= 0.1f;  // decay x0.1
    __syncthreads();
  }
}

// ---------------------------------------------------------------------------
// K3: WMMA encoder. One block (8 waves) per sample row, blockIdx.y = pass.
// Activations stored POSITION-MAJOR in LDS so B fragments are single v16h
// (2x ds_load_b128) loads and conv2 epilogue is one v8h (ds_store_b128).
// ---------------------------------------------------------------------------
struct EncArgs {
  const float* pts[2];
  const int*   starts[2];
  const float* prm[2][18];  // c1w,c1b,n1g,n1b,n1m,n1v,c2w,c2b,n2g..n2v,c3w,c3b,n3g..n3v
  float*       out[4];
};

__global__ __launch_bounds__(256) void encoder_kernel(EncArgs args)
{
  extern __shared__ unsigned char smem[];
  float* xs  = (float*)smem;          // 2*516
  float* sc1 = xs + 2 * 516;          // 64
  float* sh1 = sc1 + 64;              // 64
  float* sc2 = sh1 + 64;              // 128
  float* sh2 = sc2 + 128;             // 128
  float* sc3 = sh2 + 128;             // 128
  float* sh3 = sc3 + 128;             // 128
  float* w1l = sh3 + 128;             // 640
  // position-major activations: h1t[p*H1S + ci], h2t[p*H2S + co], p = l+1
  _Float16* h1t = (_Float16*)(w1l + 640);   // 514*H1S halves (32B-aligned)
  _Float16* h2t = h1t + 514 * H1S;          // 514*H2S halves

  const int tid  = threadIdx.x;
  const int r    = blockIdx.x;   // sample row = s*B + b
  const int pass = blockIdx.y;   // 0..3
  const int pi = (pass < 2) ? 0 : 1;
  const int wi = pass & 1;
  const float* pts = args.pts[wi];
  const int b = r % B_;
  const int start = args.starts[wi][r];
  const float* const* P = args.prm[pi];
  const float* c1w = P[0];  const float* c1b = P[1];
  const float* n1g = P[2];  const float* n1b = P[3];  const float* n1m = P[4];  const float* n1v = P[5];
  const float* c2w = P[6];  const float* c2b = P[7];
  const float* n2g = P[8];  const float* n2b = P[9];  const float* n2m = P[10]; const float* n2v = P[11];
  const float* c3w = P[12]; const float* c3b = P[13];
  const float* n3g = P[14]; const float* n3b = P[15]; const float* n3m = P[16]; const float* n3v = P[17];
  float* outF = args.out[pass];

  __builtin_prefetch(c2w, 0, 2);   // global_prefetch_b8
  __builtin_prefetch(c3w, 0, 2);

  // fold BN (+conv bias) into per-channel scale/shift
  for (int c = tid; c < 320; c += 256) {
    if (c < 64) {
      float sc = n1g[c] * rsqrtf(n1v[c] + 1e-5f);
      sc1[c] = sc; sh1[c] = (c1b[c] - n1m[c]) * sc + n1b[c];
    } else if (c < 192) {
      int k = c - 64;
      float sc = n2g[k] * rsqrtf(n2v[k] + 1e-5f);
      sc2[k] = sc; sh2[k] = (c2b[k] - n2m[k]) * sc + n2b[k];
    } else {
      int k = c - 192;
      float sc = n3g[k] * rsqrtf(n3v[k] + 1e-5f);
      sc3[k] = sc; sh3[k] = (c3b[k] - n3m[k]) * sc + n3b[k];
    }
  }
  for (int i = tid; i < 640; i += 256) w1l[i] = c1w[i];
  for (int i = tid; i < 2 * 516; i += 256) {
    int ci = i / 516, p = i % 516;
    int l = p - 2;  // segment-local, zero pad outside [0,512)
    xs[ci * 516 + p] = (l >= 0 && l < SEG)
        ? pts[((size_t)b * N_ + start + l) * 2 + ci] : 0.0f;
  }
  // zero the p=0 / p=513 halo rows (only first 64/128 entries are ever read)
  if (tid < 64)  { h1t[tid] = (_Float16)0.0f; h1t[513 * H1S + tid] = (_Float16)0.0f; }
  if (tid < 128) { h2t[tid] = (_Float16)0.0f; h2t[513 * H2S + tid] = (_Float16)0.0f; }
  __syncthreads();

  // conv1 (2->64, k5) + BN + relu -> h1t (position-major)
  for (int o = tid; o < 64 * 512; o += 256) {
    int co = o & 63, l = o >> 6;
    float acc = 0.0f;
    #pragma unroll
    for (int ci = 0; ci < 2; ++ci)
      #pragma unroll
      for (int t = 0; t < 5; ++t)
        acc += w1l[(co * 2 + ci) * 5 + t] * xs[ci * 516 + l + t];
    float v = fmaxf(acc * sc1[co] + sh1[co], 0.0f);
    h1t[(l + 1) * H1S + co] = (_Float16)v;
  }
  __syncthreads();

  const int wv = tid >> 5, lane = tid & 31, hh = lane >> 4, mm = lane & 15;
  const int coA = 16 * wv + mm;   // A-fragment row (C_out)

  // ---- conv2 (64->128, k3): GEMM 128 x 192 x 512, K-step 32, 6 steps ----
  // A fragment (16-bit A 16x32): lane L: M=L%16; slots j -> K=16h+j.
  // Within a K-tile the conv tap t is constant.
  v16h a2f[6];
  #pragma unroll
  for (int kk = 0; kk < 6; ++kk) {
    int t = kk >> 1;
    #pragma unroll
    for (int j = 0; j < 16; ++j) {
      int kl = 16 * hh + j;
      int ci = (kk & 1) * 32 + kl;
      a2f[kk][j] = (_Float16)c2w[(coA * 64 + ci) * 3 + t];
    }
  }
  for (int nt = 0; nt < 32; ++nt) {
    v8f acc = {};
    int col = nt * 16 + mm;
    #pragma unroll
    for (int kk = 0; kk < 6; ++kk) {
      int t = kk >> 1;
      int cibase = (kk & 1) * 32 + 16 * hh;  // B: lane L col=L%16, slots j -> K=16h+j
      // 16 contiguous halves, 32B-aligned -> vector LDS load
      v16h bf = *(const v16h*)(h1t + (col + t) * H1S + cibase);
      acc = __builtin_amdgcn_wmma_f32_16x16x32_f16(false, a2f[kk], false, bf,
                                                   (short)0, acc, false, false);
    }
    // C/D: M = rr + 8*(lane/16): the 8 slots are 8 consecutive C_out channels
    v8h hv;
    #pragma unroll
    for (int rr = 0; rr < 8; ++rr) {
      int co = 16 * wv + 8 * hh + rr;
      hv[rr] = (_Float16)fmaxf(acc[rr] * sc2[co] + sh2[co], 0.0f);
    }
    *(v8h*)(h2t + (col + 1) * H2S + 16 * wv + 8 * hh) = hv;  // 16B store
  }
  __syncthreads();

  // ---- conv3 (128->128, k3): GEMM 128 x 384 x 512, 12 K-steps ----
  v16h a3f[12];
  #pragma unroll
  for (int kk = 0; kk < 12; ++kk) {
    int t = kk >> 2;
    #pragma unroll
    for (int j = 0; j < 16; ++j) {
      int kl = 16 * hh + j;
      int ci = (kk & 3) * 32 + kl;
      a3f[kk][j] = (_Float16)c3w[(coA * 128 + ci) * 3 + t];
    }
  }
  float msum[8] = {0.f, 0.f, 0.f, 0.f, 0.f, 0.f, 0.f, 0.f};
  for (int nt = 0; nt < 32; ++nt) {
    v8f acc = {};
    int col = nt * 16 + mm;
    #pragma unroll
    for (int kk = 0; kk < 12; ++kk) {
      int t = kk >> 2;
      int cibase = (kk & 3) * 32 + 16 * hh;
      v16h bf = *(const v16h*)(h2t + (col + t) * H2S + cibase);
      acc = __builtin_amdgcn_wmma_f32_16x16x32_f16(false, a3f[kk], false, bf,
                                                   (short)0, acc, false, false);
    }
    #pragma unroll
    for (int rr = 0; rr < 8; ++rr) {
      int co = 16 * wv + rr + 8 * hh;
      msum[rr] += fmaxf(acc[rr] * sc3[co] + sh3[co], 0.0f);
    }
  }
  // mean-pool: sum over the 16 lanes (lane%16) that share each C_out row
  #pragma unroll
  for (int rr = 0; rr < 8; ++rr) {
    float v = msum[rr];
    v += __shfl_xor(v, 1);
    v += __shfl_xor(v, 2);
    v += __shfl_xor(v, 4);
    v += __shfl_xor(v, 8);
    if (mm == 0) {
      int co = 16 * wv + rr + 8 * hh;
      outF[(size_t)r * FD + co] = v * (1.0f / 512.0f);
    }
  }
}

// ---------------------------------------------------------------------------
// K4: match-feature MLP + quality assessor, one block per sample row.
// ---------------------------------------------------------------------------
__global__ __launch_bounds__(256) void head_kernel(
    const float* __restrict__ f1, const float* __restrict__ f2,
    const float* __restrict__ qf1, const float* __restrict__ qf2,
    const float* __restrict__ me1w, const float* __restrict__ me1b,
    const float* __restrict__ me2w, const float* __restrict__ me2b,
    const float* __restrict__ q1w, const float* __restrict__ q1b,
    const float* __restrict__ q2w, const float* __restrict__ q2b,
    const float* __restrict__ q3w, const float* __restrict__ q3b,
    float* __restrict__ mf, float* __restrict__ qual)
{
  __shared__ float f1v[128], f2v[128], comb[513], h1[256], q64v[64];
  __shared__ float redA[128], redB[128], redC[128];
  const int r = blockIdx.x, tid = threadIdx.x;

  if (tid < 128) {
    float a = f1[(size_t)r * FD + tid], c = f2[(size_t)r * FD + tid];
    f1v[tid] = a; f2v[tid] = c;
    redA[tid] = a * c; redB[tid] = a * a; redC[tid] = c * c;
  }
  __syncthreads();
  for (int s = 64; s > 0; s >>= 1) {
    if (tid < s) { redA[tid] += redA[tid + s]; redB[tid] += redB[tid + s]; redC[tid] += redC[tid + s]; }
    __syncthreads();
  }
  if (tid == 0)
    comb[512] = redA[0] / fmaxf(sqrtf(redB[0]) * sqrtf(redC[0]), 1e-8f);
  if (tid < 128) {
    comb[tid]       = f1v[tid];
    comb[128 + tid] = f2v[tid];
    comb[256 + tid] = f1v[tid] - f2v[tid];
    comb[384 + tid] = f1v[tid] * f2v[tid];
  }
  __syncthreads();
  { // me1: 513 -> 256
    float acc = me1b[tid];
    for (int i = 0; i < 513; ++i) acc += comb[i] * me1w[i * 256 + tid];
    h1[tid] = fmaxf(acc, 0.0f);
  }
  __syncthreads();
  if (tid < 128) { // me2: 256 -> 128
    float acc = me2b[tid];
    for (int i = 0; i < 256; ++i) acc += h1[i] * me2w[i * 128 + tid];
    mf[(size_t)r * FD + tid] = fmaxf(acc, 0.0f);
  }
  __syncthreads();
  if (tid < 128) {  // reuse comb as concat(q1,q2)
    comb[tid]       = qf1[(size_t)r * FD + tid];
    comb[128 + tid] = qf2[(size_t)r * FD + tid];
  }
  __syncthreads();
  if (tid < 128) { // q1: 256 -> 128
    float acc = q1b[tid];
    for (int i = 0; i < 256; ++i) acc += comb[i] * q1w[i * 128 + tid];
    h1[tid] = fmaxf(acc, 0.0f);
  }
  __syncthreads();
  if (tid < 64) { // q2: 128 -> 64
    float acc = q2b[tid];
    for (int i = 0; i < 128; ++i) acc += h1[i] * q2w[i * 64 + tid];
    q64v[tid] = fmaxf(acc, 0.0f);
  }
  __syncthreads();
  if (tid == 0) { // q3: 64 -> 1, sigmoid
    float acc = q3b[0];
    for (int i = 0; i < 64; ++i) acc += q64v[i] * q3w[i];
    qual[r] = 1.0f / (1.0f + __expf(-acc));
  }
}

// ---------------------------------------------------------------------------
// K5: attention aggregation over S samples + classifier. One block per batch.
// ---------------------------------------------------------------------------
__global__ __launch_bounds__(128) void attn_final_kernel(
    const float* __restrict__ mf, const float* __restrict__ qual,
    const float* __restrict__ a1w, const float* __restrict__ a1b,
    const float* __restrict__ a2w, const float* __restrict__ a2b,
    const float* __restrict__ a3w, const float* __restrict__ a3b,
    const float* __restrict__ c1w, const float* __restrict__ c1b,
    const float* __restrict__ c2w, const float* __restrict__ c2b,
    const float* __restrict__ c3w, const float* __restrict__ c3b,
    float* __restrict__ out)
{
  __shared__ float sfall[S_ * 128];
  __shared__ float wv[S_], t128[128], t64[64], agg[128];
  const int b = blockIdx.x, tid = threadIdx.x;

  for (int s = 0; s < S_; ++s)
    sfall[s * 128 + tid] = mf[(size_t)(s * B_ + b) * FD + tid];
  __syncthreads();

  for (int s = 0; s < S_; ++s) {
    float qs = qual[s * B_ + b];
    float acc = a1b[tid];
    for (int i = 0; i < 128; ++i) acc += sfall[s * 128 + i] * a1w[i * 128 + tid];
    acc += qs * a1w[128 * 128 + tid];
    t128[tid] = fmaxf(acc, 0.0f);
    __syncthreads();
    if (tid < 64) {
      float a = a2b[tid];
      for (int i = 0; i < 128; ++i) a += t128[i] * a2w[i * 64 + tid];
      t64[tid] = fmaxf(a, 0.0f);
    }
    __syncthreads();
    if (tid == 0) {
      float a = a3b[0];
      for (int i = 0; i < 64; ++i) a += t64[i] * a3w[i];
      wv[s] = a;
    }
    __syncthreads();
  }
  if (tid == 0) {
    float m = -1e30f;
    for (int s = 0; s < S_; ++s) m = fmaxf(m, wv[s]);
    float tot = 0.0f;
    for (int s = 0; s < S_; ++s) { wv[s] = __expf(wv[s] - m); tot += wv[s]; }
    float inv = 1.0f / tot;
    for (int s = 0; s < S_; ++s) wv[s] *= inv;
  }
  __syncthreads();
  {
    float g = 0.0f;
    for (int s = 0; s < S_; ++s) g += wv[s] * sfall[s * 128 + tid];
    agg[tid] = g;
  }
  __syncthreads();
  {
    float acc = c1b[tid];
    for (int i = 0; i < 128; ++i) acc += agg[i] * c1w[i * 128 + tid];
    t128[tid] = fmaxf(acc, 0.0f);
  }
  __syncthreads();
  if (tid < 64) {
    float a = c2b[tid];
    for (int i = 0; i < 128; ++i) a += t128[i] * c2w[i * 64 + tid];
    t64[tid] = fmaxf(a, 0.0f);
  }
  __syncthreads();
  if (tid == 0) {
    float a = c3b[0];
    for (int i = 0; i < 64; ++i) a += t64[i] * c3w[i];
    out[b] = a;
  }
}

// ---------------------------------------------------------------------------
// d_in layout (setup_inputs dict order, leaves in insertion order):
//  0 points1, 1 points2,
//  2..7   imp:  c1.w,c1.b, c2.w,c2.b, c3.w,c3.b
//  8..25  seg:  c1.w,c1.b, n1.g,n1.b,n1.m,n1.v, c2.w,c2.b, n2.g..n2.v,
//               c3.w,c3.b, n3.g..n3.v
//  26..43 qa:   (same 18-leaf pattern)
//  44..65 me1.w/b, me2.w/b, q1.w/b, q2.w/b, q3.w/b, a1.w/b, a2.w/b, a3.w/b,
//               c1.w/b, c2.w/b, c3.w/b
// ---------------------------------------------------------------------------
extern "C" void kernel_launch(void* const* d_in, const int* in_sizes, int n_in,
                              void* d_out, int out_size, void* d_ws, size_t ws_size,
                              hipStream_t stream)
{
  (void)in_sizes; (void)n_in; (void)out_size; (void)ws_size;
  const float* pts1 = (const float*)d_in[0];
  const float* pts2 = (const float*)d_in[1];
  auto F = [&](int i) { return (const float*)d_in[i]; };

  float* ws    = (float*)d_ws;
  float* comb1 = ws;                                  // 32*16384
  float* comb2 = comb1 + (size_t)B_ * N_;             // 32*16384
  int*   st1   = (int*)(comb2 + (size_t)B_ * N_);     // 1024
  int*   st2   = st1 + SB;                            // 1024
  float* f1    = (float*)(st2 + SB);                  // 1024*128
  float* f2    = f1 + (size_t)SB * FD;
  float* qf1   = f2 + (size_t)SB * FD;
  float* qf2   = qf1 + (size_t)SB * FD;
  float* mfb   = qf2 + (size_t)SB * FD;
  float* qualb = mfb + (size_t)SB * FD;               // 1024

  // K1: importance + curvature -> comb rows
  size_t lds1 = (size_t)(2 * 518 + 640 + 64 + 6144 + 32 + 32) * sizeof(float)
              + (size_t)64 * 514 * sizeof(_Float16);
  imp_comb_kernel<<<dim3(32, B_, 2), 256, lds1, stream>>>(
      pts1, pts2, F(2), F(3), F(4), F(5), F(6), F(7), comb1, comb2);

  // K2: 32 sequential multinomial draws per (batch, which)
  size_t lds2 = (size_t)(N_ + 256 + 8) * sizeof(float);
  sampling_kernel<<<dim3(B_, 2), 256, lds2, stream>>>(comb1, comb2, st1, st2);

  // K3: WMMA conv encoders (seg & qa on both point sets)
  EncArgs ea;
  ea.pts[0] = pts1; ea.pts[1] = pts2;
  ea.starts[0] = st1; ea.starts[1] = st2;
  for (int k = 0; k < 18; ++k) { ea.prm[0][k] = F(8 + k); ea.prm[1][k] = F(26 + k); }
  ea.out[0] = f1; ea.out[1] = f2; ea.out[2] = qf1; ea.out[3] = qf2;
  size_t lds3 = (size_t)(2 * 516 + 64 * 2 + 128 * 4 + 640) * sizeof(float)
              + (size_t)(514 * H1S + 514 * H2S) * sizeof(_Float16);
  encoder_kernel<<<dim3(SB, 4), 256, lds3, stream>>>(ea);

  // K4: match features + quality
  head_kernel<<<SB, 256, 0, stream>>>(
      f1, f2, qf1, qf2,
      F(44), F(45), F(46), F(47),
      F(48), F(49), F(50), F(51), F(52), F(53),
      mfb, qualb);

  // K5: attention aggregation + classifier
  attn_final_kernel<<<B_, 128, 0, stream>>>(
      mfb, qualb,
      F(54), F(55), F(56), F(57), F(58), F(59),
      F(60), F(61), F(62), F(63), F(64), F(65),
      (float*)d_out);
}
